// CausalSelfAttention_37838661878632
// MI455X (gfx1250) — matrix operand order repaired and checked by reading, not compiled, not run on hardware
//
#include <hip/hip_runtime.h>
#include <hip/hip_fp16.h>

typedef __attribute__((ext_vector_type(16))) _Float16 v16h;
typedef __attribute__((ext_vector_type(8)))  _Float16 v8h;
typedef __attribute__((ext_vector_type(8)))  float    v8f;
typedef __attribute__((ext_vector_type(4)))  float    v4f;
typedef __attribute__((ext_vector_type(4)))  unsigned int u32x4;
typedef __attribute__((ext_vector_type(8)))  int      i32x8;
typedef __attribute__((ext_vector_type(4)))  int      i32x4;

#define LDT 72          // 64 K-halves + 8 pad (16B-aligned rows, spreads banks)
#define KLD 136         // 128 + 8 pad for attention K tile (matches TDM pad)
#define VLD 72          // 64 + 8 pad for transposed V tile / P tile

__device__ inline v8f vzero() {
    v8f z;
#pragma unroll
    for (int i = 0; i < 8; ++i) z[i] = 0.0f;
    return z;
}

__device__ inline v8f wmma16(v16h a, v16h b, v8f c) {
    return __builtin_amdgcn_wmma_f32_16x16x32_f16(false, a, false, b, (short)0, c,
                                                  false, false);
}

// A fragment (16x32 f16): lane<16 -> row=lane, K = ks+0..7 & ks+16..23
//                         lane>=16 -> row=lane-16, K = ks+8..15 & ks+24..31
__device__ inline v16h lds_afrag(const _Float16* base, int ld, int lane, int ks) {
    int r = lane & 15;
    int ko = (lane >> 4) * 8;
    const _Float16* p = base + r * ld + ks + ko;
    v8h lo = *(const v8h*)p;
    v8h hi = *(const v8h*)(p + 16);
    return __builtin_shufflevector(lo, hi, 0, 1, 2, 3, 4, 5, 6, 7,
                                   8, 9, 10, 11, 12, 13, 14, 15);
}

// B fragment (32x16 f16), tile stored [N][K]: lane<16 -> col=lane, K=ks+0..7 & ks+16..23
//                                             lane>=16 -> col=lane-16, K=ks+8..15 & ks+24..31
__device__ inline v16h lds_bfrag(const _Float16* base, int ld, int lane, int ks) {
    int c = lane & 15;
    int ko = (lane >> 4) * 8;
    const _Float16* p = base + c * ld + ks + ko;
    v8h lo = *(const v8h*)p;
    v8h hi = *(const v8h*)(p + 16);
    return __builtin_shufflevector(lo, hi, 0, 1, 2, 3, 4, 5, 6, 7,
                                   8, 9, 10, 11, 12, 13, 14, 15);
}

// ---- tile stagers (register prefetch -> LDS) -------------------------------
__device__ inline void ldg_a32(const float* x, int ldx, int m0, int kt, int tid,
                               v4f* ra) {
    const int row = tid >> 2;
    const int colq = (tid & 3) * 16;
#pragma unroll
    for (int rep = 0; rep < 2; ++rep) {
        const float* src = x + (size_t)(m0 + row + rep * 64) * ldx + kt + colq;
#pragma unroll
        for (int i = 0; i < 4; ++i) ra[rep * 4 + i] = *(const v4f*)(src + i * 4);
    }
}
__device__ inline void sts_a32(_Float16* At, int tid, const v4f* ra) {
    const int row = tid >> 2;
    const int colq = (tid & 3) * 16;
#pragma unroll
    for (int rep = 0; rep < 2; ++rep) {
        _Float16* dst = At + (row + rep * 64) * LDT + colq;
#pragma unroll
        for (int i = 0; i < 4; ++i)
#pragma unroll
            for (int j = 0; j < 4; ++j) dst[i * 4 + j] = (_Float16)ra[rep * 4 + i][j];
    }
}
__device__ inline void ldg_b32(const float* w, int ldw, int n0, int kt, int tid,
                               v4f* rb) {
    const int krow = tid >> 3;
    const int nq = (tid & 7) * 16;
#pragma unroll
    for (int rep = 0; rep < 2; ++rep) {
        const float* src = w + (size_t)(kt + krow + rep * 32) * ldw + n0 + nq;
#pragma unroll
        for (int i = 0; i < 4; ++i) rb[rep * 4 + i] = *(const v4f*)(src + i * 4);
    }
}
__device__ inline void sts_b32(_Float16* Bt, int tid, const v4f* rb) {
    const int krow = tid >> 3;
    const int nq = (tid & 7) * 16;
#pragma unroll
    for (int rep = 0; rep < 2; ++rep) {
        const int k = krow + rep * 32;
#pragma unroll
        for (int i = 0; i < 4; ++i)
#pragma unroll
            for (int j = 0; j < 4; ++j)
                Bt[(nq + i * 4 + j) * LDT + k] = (_Float16)rb[rep * 4 + i][j];
    }
}
__device__ inline void ldg_a16(const _Float16* y, int m0, int kt, int tid, v8h* ra) {
    const int row = tid >> 2;
    const int colq = (tid & 3) * 16;
#pragma unroll
    for (int rep = 0; rep < 2; ++rep) {
        const _Float16* src = y + (size_t)(m0 + row + rep * 64) * 2048 + kt + colq;
        ra[rep * 2 + 0] = *(const v8h*)src;
        ra[rep * 2 + 1] = *(const v8h*)(src + 8);
    }
}
__device__ inline void sts_a16(_Float16* At, int tid, const v8h* ra) {
    const int row = tid >> 2;
    const int colq = (tid & 3) * 16;
#pragma unroll
    for (int rep = 0; rep < 2; ++rep) {
        _Float16* dst = At + (row + rep * 64) * LDT + colq;
        *(v8h*)dst = ra[rep * 2 + 0];
        *(v8h*)(dst + 8) = ra[rep * 2 + 1];
    }
}

__device__ inline void mma_tile(const _Float16* At, const _Float16* Bt, int waveM,
                                int waveN, int lane, v8f (&acc)[2][4]) {
#pragma unroll
    for (int ks = 0; ks < 64; ks += 32) {
        v16h af[2], bf[4];
#pragma unroll
        for (int mi = 0; mi < 2; ++mi)
            af[mi] = lds_afrag(At + (waveM * 32 + mi * 16) * LDT, LDT, lane, ks);
#pragma unroll
        for (int ni = 0; ni < 4; ++ni)
            bf[ni] = lds_bfrag(Bt + (waveN * 64 + ni * 16) * LDT, LDT, lane, ks);
#pragma unroll
        for (int mi = 0; mi < 2; ++mi)
#pragma unroll
            for (int ni = 0; ni < 4; ++ni)
                acc[mi][ni] = wmma16(af[mi], bf[ni], acc[mi][ni]);
    }
}

// ---------------------------------------------------------------------------
// Kernel A: qkv = x @ wqkv  (M=8192, K=2048, N=6144), fused RoPE epilogue,
// scatter q,k,v (f16) into [B*H][T][D] workspace. Double-buffered via regs.
// ---------------------------------------------------------------------------
__global__ __launch_bounds__(256) void qkv_rope_kernel(
    const float* __restrict__ x, const float* __restrict__ wqkv,
    const float* __restrict__ rope_cos, const float* __restrict__ rope_sin,
    _Float16* __restrict__ qws, _Float16* __restrict__ kws,
    _Float16* __restrict__ vws) {
    __shared__ _Float16 At[128 * LDT];
    __shared__ _Float16 Bt[128 * LDT];

    const int tid = threadIdx.x;
    const int lane = tid & 31;
    const int wid = tid >> 5;
    const int waveM = wid >> 1;
    const int waveN = wid & 1;
    const int m0 = blockIdx.y * 128;
    const int n0 = blockIdx.x * 128;

    v8f acc[2][4];
#pragma unroll
    for (int mi = 0; mi < 2; ++mi)
#pragma unroll
        for (int ni = 0; ni < 4; ++ni) acc[mi][ni] = vzero();

    v4f ra[8], rb[8];
    ldg_a32(x, 2048, m0, 0, tid, ra);
    ldg_b32(wqkv, 6144, n0, 0, tid, rb);
    sts_a32(At, tid, ra);
    sts_b32(Bt, tid, rb);
    __syncthreads();
    for (int kt = 64; kt < 2048; kt += 64) {
        ldg_a32(x, 2048, m0, kt, tid, ra);       // prefetch next tile
        ldg_b32(wqkv, 6144, n0, kt, tid, rb);
        mma_tile(At, Bt, waveM, waveN, lane, acc);  // overlaps with loads
        __syncthreads();
        sts_a32(At, tid, ra);
        sts_b32(Bt, tid, rb);
        __syncthreads();
    }
    mma_tile(At, Bt, waveM, waveN, lane, acc);

    // --- Epilogue: RoPE on q/k regions, scatter to head-major f16 ws ---
    const int rb2 = (lane >> 4) ? 8 : 0;
#pragma unroll
    for (int mi = 0; mi < 2; ++mi) {
#pragma unroll
        for (int ni = 0; ni < 4; ++ni) {
            const int colbase = n0 + waveN * 64 + ni * 16;
            const int region = colbase >> 11;  // 0=q 1=k 2=v (tile-uniform)
            const int col = colbase + (lane & 15);
            const int rowbase = m0 + waveM * 32 + mi * 16 + rb2;
            const int cc = col & 2047;
            const int h = cc >> 7;
            const int d = cc & 127;
#pragma unroll
            for (int j = 0; j < 8; ++j) {
                float v = acc[mi][ni][j];
                const int row = rowbase + j;
                const int t = row & 2047;
                const int b = row >> 11;
                float outv = v;
                if (region < 2) {  // wave-uniform branch
                    float part = __shfl_xor(v, 1);
                    const int d2 = d >> 1;
                    float c = rope_cos[t * 64 + d2];
                    float s = rope_sin[t * 64 + d2];
                    outv = (d & 1) ? (part * s + v * c) : (v * c - part * s);
                }
                const size_t addr = ((size_t)((b * 16 + h) * 2048 + t)) * 128 + d;
                const _Float16 hv = (_Float16)outv;
                if (region == 0)      qws[addr] = hv;
                else if (region == 1) kws[addr] = hv;
                else                  vws[addr] = hv;
            }
        }
    }
}

// ---------------------------------------------------------------------------
// Kernel B: causal flash attention per (b,h). Block = 128 query rows,
// wave = 16 query rows; KV tiles of 64 through LDS. K tile arrives via the
// Tensor Data Mover (async DMA, LDS padding done by the TDM descriptor).
// ---------------------------------------------------------------------------
__global__ __launch_bounds__(256) void attn_kernel(
    const _Float16* __restrict__ qws, const _Float16* __restrict__ kws,
    const _Float16* __restrict__ vws, _Float16* __restrict__ yws) {
    __shared__ _Float16 Kt[64 * KLD];     // [kv][d], stride 136 (TDM-padded)
    __shared__ _Float16 Vt[128 * VLD];    // [d][kv] (transposed)
    __shared__ _Float16 Pt[8][16 * VLD];  // per-wave P restage

    const int tid = threadIdx.x;
    const int lane = tid & 31;
    const int wid = tid >> 5;
    const int bh = blockIdx.y;
    const int q0 = blockIdx.x * 128;
    const size_t base = (size_t)bh * 2048 * 128;
    const int qrow_w = q0 + wid * 16;
    const int rb = (lane >> 4) ? 8 : 0;

    // LDS byte offset of Kt (generic shared pointer keeps LDS offset in addr[31:0])
    const unsigned kt_lds = (unsigned)(unsigned long long)(const void*)&Kt[0];

    // Q fragments (registers, pre-scaled by D^-0.5)
    v16h qf[4];
    {
        const int r = lane & 15;
        const int ko = (lane >> 4) * 8;
        const _Float16* qp = qws + base + (size_t)(qrow_w + r) * 128;
#pragma unroll
        for (int ksd = 0; ksd < 4; ++ksd) {
            const _Float16* p = qp + ksd * 32 + ko;
            v8h lo = *(const v8h*)p;
            v8h hi = *(const v8h*)(p + 16);
            v16h f = __builtin_shufflevector(lo, hi, 0, 1, 2, 3, 4, 5, 6, 7,
                                             8, 9, 10, 11, 12, 13, 14, 15);
#pragma unroll
            for (int i = 0; i < 16; ++i)
                f[i] = f[i] * (_Float16)0.08838834764831845f;
            qf[ksd] = f;
        }
    }

    v8f oacc[8];
#pragma unroll
    for (int ds = 0; ds < 8; ++ds) oacc[ds] = vzero();
    float mrow[8], lrow[8];
#pragma unroll
    for (int j = 0; j < 8; ++j) { mrow[j] = -1e30f; lrow[j] = 0.0f; }

    const int nkv = (q0 >> 6) + 2;
    for (int kvt = 0; kvt < nkv; ++kvt) {
        const int kv0 = kvt * 64;
        __syncthreads();

        // --- K tile via Tensor Data Mover: 64x128 f16, pad 4 DW per 64 DW ---
        if (wid == 0) {
            const unsigned long long ga =
                (unsigned long long)(const void*)(kws + base + (size_t)kv0 * 128);
            u32x4 g0 = {1u,                                   // count=1
                        kt_lds,                               // lds_addr (bytes)
                        (unsigned)(ga & 0xffffffffu),
                        (unsigned)((ga >> 32) & 0x1ffffffu) | (2u << 30)};  // type=2
            i32x8 g1 = {(int)((1u << 16) |       // data_size = 2B
                              (1u << 20) |       // pad_enable
                              (5u << 22) |       // pad_interval: 64 DWORDs
                              (3u << 25)),       // pad_amount:   4 DWORDs
                        (int)(128u << 16),       // tensor_dim0 = 128
                        (int)(2048u << 16),      // tensor_dim1 = 2048
                        (int)(128u << 16),       // tile_dim0 = 128
                        64,                      // tile_dim1 = 64, tile_dim2 = 0
                        128,                     // tensor_dim0_stride = 128 (lo)
                        0, 0};                   // stride hi / dim1_stride = 0
            i32x4 gz4 = {0, 0, 0, 0};
#if defined(__clang_major__) && (__clang_major__ >= 23)
            i32x8 gz8 = {0, 0, 0, 0, 0, 0, 0, 0};
            __builtin_amdgcn_tensor_load_to_lds(g0, g1, gz4, gz4, gz8, 0);
#else
            __builtin_amdgcn_tensor_load_to_lds(g0, g1, gz4, gz4, 0);
#endif
        }

        // --- V tile -> LDS transposed [d][kv] (overlaps with TDM) ---
        {
            const int row = tid >> 2;
            const int seg = (tid & 3) * 32;
            const _Float16* src = vws + base + (size_t)(kv0 + row) * 128 + seg;
#pragma unroll
            for (int i = 0; i < 32; i += 8) {
                v8h vv = *(const v8h*)(src + i);
#pragma unroll
                for (int j = 0; j < 8; ++j) Vt[(seg + i + j) * VLD + row] = vv[j];
            }
        }
        if (wid == 0) __builtin_amdgcn_s_wait_tensorcnt(0);
        __syncthreads();

        // S = Q @ K^T  (per wave: 16 x 64)
        v8f sacc[4];
#pragma unroll
        for (int ns = 0; ns < 4; ++ns) sacc[ns] = vzero();
#pragma unroll
        for (int ns = 0; ns < 4; ++ns)
#pragma unroll
            for (int ksd = 0; ksd < 4; ++ksd)
                sacc[ns] = wmma16(qf[ksd],
                                  lds_bfrag(Kt + (ns * 16) * KLD, KLD, lane, ksd * 32),
                                  sacc[ns]);

        // causal mask
#pragma unroll
        for (int ns = 0; ns < 4; ++ns) {
            const int coln = kv0 + ns * 16 + (lane & 15);
#pragma unroll
            for (int j = 0; j < 8; ++j) {
                const int rown = qrow_w + j + rb;
                sacc[ns][j] = (coln > rown) ? -1e30f : sacc[ns][j];
            }
        }

        // online softmax (row reductions across 16-lane groups)
#pragma unroll
        for (int j = 0; j < 8; ++j) {
            float vmax = fmaxf(fmaxf(sacc[0][j], sacc[1][j]),
                               fmaxf(sacc[2][j], sacc[3][j]));
#pragma unroll
            for (int off = 1; off < 16; off <<= 1)
                vmax = fmaxf(vmax, __shfl_xor(vmax, off));
            const float mnew = fmaxf(mrow[j], vmax);
            const float alpha = __expf(mrow[j] - mnew);
            mrow[j] = mnew;
            float rs = 0.0f;
#pragma unroll
            for (int ns = 0; ns < 4; ++ns) {
                sacc[ns][j] = __expf(sacc[ns][j] - mnew);
                rs += sacc[ns][j];
            }
#pragma unroll
            for (int off = 1; off < 16; off <<= 1) rs += __shfl_xor(rs, off);
            lrow[j] = lrow[j] * alpha + rs;
#pragma unroll
            for (int ds = 0; ds < 8; ++ds) oacc[ds][j] *= alpha;
        }

        // restage P (C-layout -> A-layout) through per-wave LDS
        _Float16* pw = Pt[wid];
#pragma unroll
        for (int ns = 0; ns < 4; ++ns)
#pragma unroll
            for (int j = 0; j < 8; ++j)
                pw[(j + rb) * VLD + ns * 16 + (lane & 15)] = (_Float16)sacc[ns][j];
        __asm__ volatile("s_wait_dscnt 0" ::: "memory");  // same-wave LDS RAW

        // O += P @ V
#pragma unroll
        for (int ds = 0; ds < 8; ++ds)
#pragma unroll
            for (int ks2 = 0; ks2 < 64; ks2 += 32)
                oacc[ds] = wmma16(lds_afrag(pw, VLD, lane, ks2),
                                  lds_bfrag(Vt + (ds * 16) * VLD, VLD, lane, ks2),
                                  oacc[ds]);
    }

    // epilogue: normalize, store y (f16) in [b][t][h*128+d]
    const int b = bh >> 4, h = bh & 15;
#pragma unroll
    for (int j = 0; j < 8; ++j) {
        const float inv = 1.0f / lrow[j];
        const int t = qrow_w + j + rb;
#pragma unroll
        for (int ds = 0; ds < 8; ++ds) {
            const int d = ds * 16 + (lane & 15);
            yws[((size_t)(b * 2048 + t)) * 2048 + h * 128 + d] =
                (_Float16)(oacc[ds][j] * inv);
        }
    }
}

// ---------------------------------------------------------------------------
// Kernel C: out = y @ wo  (M=8192, K=2048, N=2048), f32 output, double-buffered.
// ---------------------------------------------------------------------------
__global__ __launch_bounds__(256) void proj_kernel(
    const _Float16* __restrict__ yws, const float* __restrict__ wo,
    float* __restrict__ out) {
    __shared__ _Float16 At[128 * LDT];
    __shared__ _Float16 Bt[128 * LDT];

    const int tid = threadIdx.x;
    const int lane = tid & 31;
    const int wid = tid >> 5;
    const int waveM = wid >> 1;
    const int waveN = wid & 1;
    const int m0 = blockIdx.y * 128;
    const int n0 = blockIdx.x * 128;

    v8f acc[2][4];
#pragma unroll
    for (int mi = 0; mi < 2; ++mi)
#pragma unroll
        for (int ni = 0; ni < 4; ++ni) acc[mi][ni] = vzero();

    v8h ra[4];
    v4f rb[8];
    ldg_a16(yws, m0, 0, tid, ra);
    ldg_b32(wo, 2048, n0, 0, tid, rb);
    sts_a16(At, tid, ra);
    sts_b32(Bt, tid, rb);
    __syncthreads();
    for (int kt = 64; kt < 2048; kt += 64) {
        ldg_a16(yws, m0, kt, tid, ra);
        ldg_b32(wo, 2048, n0, kt, tid, rb);
        mma_tile(At, Bt, waveM, waveN, lane, acc);
        __syncthreads();
        sts_a16(At, tid, ra);
        sts_b32(Bt, tid, rb);
        __syncthreads();
    }
    mma_tile(At, Bt, waveM, waveN, lane, acc);

    const int rb2 = (lane >> 4) ? 8 : 0;
#pragma unroll
    for (int mi = 0; mi < 2; ++mi)
#pragma unroll
        for (int ni = 0; ni < 4; ++ni) {
            const int col = n0 + waveN * 64 + ni * 16 + (lane & 15);
            const int rowbase = m0 + waveM * 32 + mi * 16 + rb2;
#pragma unroll
            for (int j = 0; j < 8; ++j)
                out[(size_t)(rowbase + j) * 2048 + col] = acc[mi][ni][j];
        }
}

// ---------------------------------------------------------------------------
extern "C" void kernel_launch(void* const* d_in, const int* in_sizes, int n_in,
                              void* d_out, int out_size, void* d_ws,
                              size_t ws_size, hipStream_t stream) {
    const float* x    = (const float*)d_in[0];
    const float* wqkv = (const float*)d_in[1];
    const float* wo   = (const float*)d_in[2];
    const float* rc   = (const float*)d_in[3];
    const float* rs   = (const float*)d_in[4];
    float* out = (float*)d_out;

    const size_t elems = (size_t)4 * 2048 * 2048;  // B*T*C
    _Float16* qws = (_Float16*)d_ws;
    _Float16* kws = qws + elems;
    _Float16* vws = kws + elems;
    _Float16* yws = vws + elems;

    dim3 blk(256);
    qkv_rope_kernel<<<dim3(48, 64), blk, 0, stream>>>(x, wqkv, rc, rs, qws, kws, vws);
    attn_kernel<<<dim3(16, 64), blk, 0, stream>>>(qws, kws, vws, yws);
    proj_kernel<<<dim3(16, 64), blk, 0, stream>>>(yws, wo, out);
}
